// RWKVAttention_88390426952204
// MI455X (gfx1250) — matrix operand order repaired and checked by reading, not compiled, not run on hardware
//
#include <hip/hip_runtime.h>
#include <hip/hip_bf16.h>

// RWKV attention for MI455X (gfx1250, wave32).
// GEMMs: split-bf16 fp32-emulation on V_WMMA_F32_16X16X32_BF16
//   (acc += Ahi*Bhi + Ahi*Blo + Alo*Bhi) with pre-split bf16 hi/lo planes
//   (token-shift mix fused into the A split). BK=64 K-chunks: 24 WMMAs per
//   wave between barriers. Double-buffered LDS staging via
//   GLOBAL_LOAD_ASYNC_TO_LDS_B128 (ASYNCcnt DMA): tile t+1 streams in while
//   tile t feeds the WMMA pipe; s_wait_asynccnt only after compute.
// WKV scan: 3-pass chunked parallel scan (32 chunks x 128 steps); pass 3
//   fuses r*wkv and the bf16 split of the output GEMM's A operand.
// Divisions use v_rcp_f32 + one Newton step instead of IEEE div expansion.

typedef __attribute__((ext_vector_type(8)))  float  v8f;
typedef __attribute__((ext_vector_type(16))) __bf16 v16bf;
typedef __attribute__((ext_vector_type(8)))  __bf16 v8bf;
typedef __attribute__((ext_vector_type(4)))  __bf16 v4bf;
typedef int v4i_g __attribute__((vector_size(16)));   // builtin's pointee type

#if defined(__has_builtin)
# if __has_builtin(__builtin_amdgcn_global_load_async_to_lds_b128)
#  define HAVE_ASYNC_LDS 1
# endif
#endif

#ifdef HAVE_ASYNC_LDS
#define ASYNC_CP16(gsrc, ldst)                                                \
    __builtin_amdgcn_global_load_async_to_lds_b128(                           \
        (__attribute__((address_space(1))) v4i_g*)(gsrc),                     \
        (__attribute__((address_space(3))) v4i_g*)(ldst), 0, 0)
#define WAIT_ASYNC() asm volatile("s_wait_asynccnt 0" ::: "memory")
#else
#define ASYNC_CP16(gsrc, ldst) (*(v8bf*)(ldst) = *(const v8bf*)(gsrc))
#define WAIT_ASYNC()
#endif

#define BM 128
#define BN 64
#define BK 64
#define KP 72        // LDS row stride (bf16): 144B rows -> 16B-aligned b128
                     // fragment loads; 36*m mod 64 distinct for m<16 -> no
                     // bank conflicts on 16-row fragment reads

#define SEQ_L 4096
#define DIM   1024
#define SCAN_C 32    // chunks per sequence
#define SCAN_T 128   // steps per chunk  (SCAN_C * SCAN_T == SEQ_L)

// v_rcp_f32 + one Newton-Raphson step: ~0.5 ulp, 3 VALU ops.
__device__ __forceinline__ float fast_rcp(float x)
{
    float r = __builtin_amdgcn_rcpf(x);
    r = __builtin_fmaf(__builtin_fmaf(-x, r, 1.0f), r, r);
    return r;
}

// ---------------------------------------------------------------------------
// Elementwise: o = mix ? (x*mix + xprev*(1-mix)) : x ; split o into bf16 hi/lo
// planes. 4 elements (one float4) per thread.
// ---------------------------------------------------------------------------
__global__ __launch_bounds__(256)
void mix_split_kernel(const float* __restrict__ X, const float* __restrict__ mix,
                      __bf16* __restrict__ hi, __bf16* __restrict__ lo, int Lseq)
{
    int idx = blockIdx.x * 256 + threadIdx.x;          // float4 index
    int e   = idx << 2;
    int gr  = e >> 10;                                 // row (D = 1024)
    int col = e & (DIM - 1);

    float4 o = *(const float4*)(X + (size_t)gr * DIM + col);
    if (mix) {
        float4 xp = make_float4(0.f, 0.f, 0.f, 0.f);
        if ((gr % Lseq) != 0)
            xp = *(const float4*)(X + (size_t)(gr - 1) * DIM + col);
        const float4 mm = *(const float4*)(mix + col);
        o.x = o.x * mm.x + xp.x * (1.f - mm.x);
        o.y = o.y * mm.y + xp.y * (1.f - mm.y);
        o.z = o.z * mm.z + xp.z * (1.f - mm.z);
        o.w = o.w * mm.w + xp.w * (1.f - mm.w);
    }
    float f[4] = {o.x, o.y, o.z, o.w};
    v4bf h, l;
    #pragma unroll
    for (int j = 0; j < 4; ++j) {
        __bf16 hh = (__bf16)f[j];
        h[j] = hh;
        l[j] = (__bf16)(f[j] - (float)hh);
    }
    *(v4bf*)(hi + (size_t)e) = h;
    *(v4bf*)(lo + (size_t)e) = l;
}

// ---------------------------------------------------------------------------
// Split-bf16 GEMM: Y[m,n] = act( sum_d A(m,d) * W[n,d] ), A/W given as bf16
// hi/lo planes. Tile 128x64 per 256-thread block; 8 waves of 32x32; per K=64
// chunk each wave issues 24 v_wmma_f32_16x16x32_bf16; double-buffered async
// LDS staging overlaps the next tile's DMA with the current tile's WMMAs.
// ---------------------------------------------------------------------------
__global__ __launch_bounds__(256)
void bf16x3_gemm(const __bf16* __restrict__ Ahi, const __bf16* __restrict__ Alo,
                 const __bf16* __restrict__ Whi, const __bf16* __restrict__ Wlo,
                 float* __restrict__ Y, int act)
{
    __shared__ __bf16 XsH[2][BM][KP], XsL[2][BM][KP];
    __shared__ __bf16 WsH[2][BN][KP], WsL[2][BN][KP];

    const int tid  = threadIdx.x;
    const int lane = tid & 31;
    const int wave = tid >> 5;
    const int wm = wave >> 1;           // 0..3 -> 32-row strip
    const int wn = wave & 1;            // 0..1 -> 32-col strip
    const int m0 = blockIdx.x * BM;
    const int n0 = blockIdx.y * BN;

    const int ml = lane & 15;
    const int h  = lane >> 4;           // half-wave select

    v8f acc[2][2] = {};

#define STAGE_TILE(buf, koff)                                                 \
    do {                                                                      \
        _Pragma("unroll")                                                     \
        for (int it = 0; it < 4; ++it) {                                      \
            int e   = tid + it * 256;       /* v8bf slot (1024/plane) */      \
            int row = e >> 3;                                                 \
            int c8  = (e & 7) << 3;                                           \
            size_t g = (size_t)(m0 + row) * DIM + (koff) + c8;                \
            ASYNC_CP16(Ahi + g, &XsH[buf][row][c8]);                          \
            ASYNC_CP16(Alo + g, &XsL[buf][row][c8]);                          \
        }                                                                     \
        _Pragma("unroll")                                                     \
        for (int it = 0; it < 2; ++it) {                                      \
            int e   = tid + it * 256;       /* v8bf slot (512/plane) */       \
            int row = e >> 3;                                                 \
            int c8  = (e & 7) << 3;                                           \
            size_t g = (size_t)(n0 + row) * DIM + (koff) + c8;                \
            ASYNC_CP16(Whi + g, &WsH[buf][row][c8]);                          \
            ASYNC_CP16(Wlo + g, &WsL[buf][row][c8]);                          \
        }                                                                     \
    } while (0)

    // Prologue: stage tile 0 into buffer 0.
    STAGE_TILE(0, 0);
    WAIT_ASYNC();
    __syncthreads();

    const int NIT = DIM / BK;           // 16 K-iterations
    for (int itK = 0; itK < NIT; ++itK) {
        const int buf = itK & 1;
        // Issue DMA for the next tile into the other buffer; it streams in
        // while this tile's 24 WMMAs execute. Safe: the barrier at the end of
        // iteration itK-1 guaranteed all waves finished reading buf^1.
        if (itK + 1 < NIT)
            STAGE_TILE(buf ^ 1, (itK + 1) * BK);

        #pragma unroll
        for (int kk = 0; kk < BK; kk += 32) {
            // ---- fragments per ISA 16-bit layouts ----
            // A 16x32: elems 0..7 = K[h*8 .. +7], elems 8..15 = K[16+h*8 ..]
            // B 32x16: elems 0..7 = K[h*8 .. +7], elems 8..15 = K[16+h*8 ..] at column ml
            v16bf aH[2], aL[2], bH[2], bL[2];
            #pragma unroll
            for (int i = 0; i < 2; ++i) {
                int r = wm * 32 + i * 16 + ml;
                v8bf p0 = *(const v8bf*)&XsH[buf][r][kk + h * 8];
                v8bf p1 = *(const v8bf*)&XsH[buf][r][kk + 16 + h * 8];
                aH[i] = __builtin_shufflevector(p0, p1, 0,1,2,3,4,5,6,7,
                                                        8,9,10,11,12,13,14,15);
                p0 = *(const v8bf*)&XsL[buf][r][kk + h * 8];
                p1 = *(const v8bf*)&XsL[buf][r][kk + 16 + h * 8];
                aL[i] = __builtin_shufflevector(p0, p1, 0,1,2,3,4,5,6,7,
                                                        8,9,10,11,12,13,14,15);
            }
            #pragma unroll
            for (int j = 0; j < 2; ++j) {
                int r = wn * 32 + j * 16 + ml;
                v8bf p0 = *(const v8bf*)&WsH[buf][r][kk + h * 8];
                v8bf p1 = *(const v8bf*)&WsH[buf][r][kk + 16 + h * 8];
                bH[j] = __builtin_shufflevector(p0, p1, 0,1,2,3,4,5,6,7,
                                                        8,9,10,11,12,13,14,15);
                p0 = *(const v8bf*)&WsL[buf][r][kk + h * 8];
                p1 = *(const v8bf*)&WsL[buf][r][kk + 16 + h * 8];
                bL[j] = __builtin_shufflevector(p0, p1, 0,1,2,3,4,5,6,7,
                                                        8,9,10,11,12,13,14,15);
            }

            #pragma unroll
            for (int i = 0; i < 2; ++i)
                #pragma unroll
                for (int j = 0; j < 2; ++j) {
                    acc[i][j] = __builtin_amdgcn_wmma_f32_16x16x32_bf16(
                        false, aH[i], false, bH[j], (short)0, acc[i][j],
                        false, false);
                    acc[i][j] = __builtin_amdgcn_wmma_f32_16x16x32_bf16(
                        false, aH[i], false, bL[j], (short)0, acc[i][j],
                        false, false);
                    acc[i][j] = __builtin_amdgcn_wmma_f32_16x16x32_bf16(
                        false, aL[i], false, bH[j], (short)0, acc[i][j],
                        false, false);
                }
        }

        // Publish the next tile: wait for this wave's DMAs, then barrier.
        WAIT_ASYNC();
        __syncthreads();
    }
#undef STAGE_TILE

    // ---- epilogue: optional fast sigmoid; C/D layout: lanes 16-31 = M+8 ----
    #pragma unroll
    for (int i = 0; i < 2; ++i)
        #pragma unroll
        for (int j = 0; j < 2; ++j)
            #pragma unroll
            for (int r = 0; r < 8; ++r) {
                int gm = m0 + wm * 32 + i * 16 + h * 8 + r;
                int gn = n0 + wn * 32 + j * 16 + ml;
                float vv = acc[i][j][r];
                if (act == 1) vv = fast_rcp(1.0f + __expf(-vv));
                Y[(size_t)gm * DIM + gn] = vv;
            }
}

// ---------------------------------------------------------------------------
// WKV chunked scan. Pass 1: per-(b,d,chunk) partial (a,s) from zero state.
// ---------------------------------------------------------------------------
__global__ __launch_bounds__(256)
void wkv_pass1(const float* __restrict__ k, const float* __restrict__ v,
               const float* __restrict__ td,
               float* __restrict__ pa, float* __restrict__ pb)
{
    int idx = blockIdx.x * 256 + threadIdx.x;      // B * SCAN_C * D
    int d = idx & (DIM - 1);
    int c = (idx >> 10) & (SCAN_C - 1);
    int b = idx >> 15;

    float decay = __expf(-__expf(td[d]));
    float a = 0.f, s = 0.f;
    size_t base = ((size_t)b * SEQ_L + (size_t)c * SCAN_T) * DIM + d;
    for (int t = 0; t < SCAN_T; ++t) {
        float kt = k[base], vt = v[base];
        float ek = __expf(kt);
        a = decay * a + ek * vt;
        s = decay * s + ek;
        base += DIM;
    }
    size_t o = ((size_t)b * SCAN_C + c) * DIM + d;
    pa[o] = a;
    pb[o] = s;
}

// Pass 2: per-(b,d) combine over 32 chunks -> exact chunk starting states.
__global__ __launch_bounds__(256)
void wkv_pass2(const float* __restrict__ pa, const float* __restrict__ pb,
               const float* __restrict__ td,
               float* __restrict__ sa, float* __restrict__ sb)
{
    int idx = blockIdx.x * 256 + threadIdx.x;      // B * D
    int d = idx & (DIM - 1);
    int b = idx >> 10;
    float decayT = __expf(-__expf(td[d]) * (float)SCAN_T);   // decay^SCAN_T
    float a = 0.f, s = 0.f;
    for (int c = 0; c < SCAN_C; ++c) {
        size_t o = ((size_t)b * SCAN_C + c) * DIM + d;
        sa[o] = a;
        sb[o] = s;
        a = decayT * a + pa[o];
        s = decayT * s + pb[o];
    }
}

// Pass 3: replay each chunk from its exact start state (reference math),
// fuse r*wkv and the bf16 hi/lo split of the output GEMM's A operand.
__global__ __launch_bounds__(256)
void wkv_pass3(const float* __restrict__ k, const float* __restrict__ v,
               const float* __restrict__ r,
               const float* __restrict__ sa, const float* __restrict__ sb,
               const float* __restrict__ td, const float* __restrict__ tf,
               __bf16* __restrict__ ahi, __bf16* __restrict__ alo)
{
    int idx = blockIdx.x * 256 + threadIdx.x;      // B * SCAN_C * D
    int d = idx & (DIM - 1);
    int c = (idx >> 10) & (SCAN_C - 1);
    int b = idx >> 15;

    float decay = __expf(-__expf(td[d]));
    float u = tf[d];
    size_t so = ((size_t)b * SCAN_C + c) * DIM + d;
    float a = sa[so], s = sb[so];

    size_t base = ((size_t)b * SEQ_L + (size_t)c * SCAN_T) * DIM + d;
    for (int t = 0; t < SCAN_T; ++t) {
        float kt = k[base], vt = v[base];
        float eku = __expf(u + kt);
        float wkv = (a + eku * vt) * fast_rcp(fmaxf(s + eku, 1e-6f));
        float ek  = __expf(kt);
        a = decay * a + ek * vt;
        s = decay * s + ek;
        float rw = r[base] * wkv;
        __bf16 hh = (__bf16)rw;
        ahi[base] = hh;
        alo[base] = (__bf16)(rw - (float)hh);
        base += DIM;
    }
}

// ---------------------------------------------------------------------------
extern "C" void kernel_launch(void* const* d_in, const int* in_sizes, int n_in,
                              void* d_out, int out_size, void* d_ws, size_t ws_size,
                              hipStream_t stream)
{
    const float* x  = (const float*)d_in[0];
    const float* td = (const float*)d_in[1];
    const float* tf = (const float*)d_in[2];
    const float* mk = (const float*)d_in[3];
    const float* mv = (const float*)d_in[4];
    const float* mr = (const float*)d_in[5];
    const float* Wk = (const float*)d_in[6];
    const float* Wv = (const float*)d_in[7];
    const float* Wr = (const float*)d_in[8];
    const float* Wo = (const float*)d_in[9];
    float* out = (float*)d_out;

    const int Bb = 4, M = Bb * SEQ_L;
    const size_t plane = (size_t)M * DIM;          // 16M elements

    // Workspace layout (~274 MB):
    char* p = (char*)d_ws;
    float* kbuf = (float*)p;                 p += plane * 4;   // 64 MB
    float* vbuf = (float*)p;                 p += plane * 4;   // 64 MB
    float* rbuf = (float*)p;                 p += plane * 4;   // 64 MB
    __bf16* AH  = (__bf16*)p;                p += plane * 2;   // 32 MB
    __bf16* AL  = (__bf16*)p;                p += plane * 2;   // 32 MB
    const size_t wpl = (size_t)DIM * DIM;                      // 1M elements
    __bf16* WH[4]; __bf16* WL[4];
    for (int i = 0; i < 4; ++i) {
        WH[i] = (__bf16*)p; p += wpl * 2;                      // 2 MB each
        WL[i] = (__bf16*)p; p += wpl * 2;
    }
    const size_t spl = (size_t)Bb * SCAN_C * DIM;              // 128K elements
    float* pa = (float*)p; p += spl * 4;
    float* pb = (float*)p; p += spl * 4;
    float* sa = (float*)p; p += spl * 4;
    float* sb = (float*)p; p += spl * 4;

    const dim3 blk(256);
    const dim3 gGemm(M / BM, DIM / BN);
    const int gSplitA = (int)(plane / 4 / 256);    // 16384 blocks
    const int gSplitW = (int)(wpl / 4 / 256);      // 1024 blocks
    const int gScan   = Bb * SCAN_C * DIM / 256;   // 512 blocks
    const int gComb   = Bb * DIM / 256;            // 16 blocks

    // Split all weight matrices once (reused by all 128 M-tiles).
    const float* Ws[4] = {Wk, Wv, Wr, Wo};
    for (int i = 0; i < 4; ++i)
        mix_split_kernel<<<gSplitW, blk, 0, stream>>>(Ws[i], nullptr,
                                                      WH[i], WL[i], 1);

    // k = xk @ Wk^T
    mix_split_kernel<<<gSplitA, blk, 0, stream>>>(x, mk, AH, AL, SEQ_L);
    bf16x3_gemm<<<gGemm, blk, 0, stream>>>(AH, AL, WH[0], WL[0], kbuf, 0);
    // v = xv @ Wv^T
    mix_split_kernel<<<gSplitA, blk, 0, stream>>>(x, mv, AH, AL, SEQ_L);
    bf16x3_gemm<<<gGemm, blk, 0, stream>>>(AH, AL, WH[1], WL[1], vbuf, 0);
    // r = sigmoid(xr @ Wr^T)
    mix_split_kernel<<<gSplitA, blk, 0, stream>>>(x, mr, AH, AL, SEQ_L);
    bf16x3_gemm<<<gGemm, blk, 0, stream>>>(AH, AL, WH[2], WL[2], rbuf, 1);

    // Chunked WKV scan; pass 3 writes bf16 split of (r*wkv) into AH/AL.
    wkv_pass1<<<gScan, blk, 0, stream>>>(kbuf, vbuf, td, pa, pb);
    wkv_pass2<<<gComb, blk, 0, stream>>>(pa, pb, td, sa, sb);
    wkv_pass3<<<gScan, blk, 0, stream>>>(kbuf, vbuf, rbuf, sa, sb, td, tf,
                                         AH, AL);

    // out = (r*wkv) @ Wo^T
    bf16x3_gemm<<<gGemm, blk, 0, stream>>>(AH, AL, WH[3], WL[3], out, 0);
}
